// Self_Consistency_38603166056891
// MI455X (gfx1250) — matrix operand order repaired and checked by reading, not compiled, not run on hardware
//
#include <hip/hip_runtime.h>
#include <hip/hip_bf16.h>
#include <math.h>

typedef __attribute__((ext_vector_type(16))) _Float16 v16h;
typedef __attribute__((ext_vector_type(8)))  _Float16 v8h;
typedef __attribute__((ext_vector_type(8)))  float    v8f;

// ---------------------------------------------------------------------------
// Fragment loaders (layouts per CDNA5 ISA 7.12.2, wave32)
//   A (16x32 f16): lane<16 -> row M=lane, halves0..7 = K k0+hi*8..+7,
//                  halves8..15 = K k0+16+hi*8..+7   (hi = lane>>4)
//   B (32x16 f16): lane%16 -> col N, halves0..7 = K k0+hi*8..+7,
//                  halves8..15 = K k0+16+hi*8..+7   (same as A)
//   C/D (16x16 f32): lane%16 -> col N, vgpr r -> row M = hi*8 + r
// ---------------------------------------------------------------------------
__device__ __forceinline__ v16h ldA(const _Float16* __restrict__ p) {
  // p = row_base + k0 + hi*8 ; two 16B contiguous loads
  v8h lo = *(const v8h*)(p);
  v8h hh = *(const v8h*)(p + 16);
  v16h r;
#pragma unroll
  for (int i = 0; i < 8; ++i) { r[i] = lo[i]; r[8 + i] = hh[i]; }
  return r;
}

__device__ __forceinline__ v16h ldB(const _Float16* __restrict__ p) {
  // p = col_base + k0 + hi*8 ; two 16B contiguous loads
  v8h lo = *(const v8h*)(p);
  v8h hh = *(const v8h*)(p + 16);
  v16h r;
#pragma unroll
  for (int i = 0; i < 8; ++i) { r[i] = lo[i]; r[8 + i] = hh[i]; }
  return r;
}

__device__ __forceinline__ v8f wmma_f16(v16h a, v16h b, v8f c) {
  return __builtin_amdgcn_wmma_f32_16x16x32_f16(false, a, false, b, (short)0, c,
                                                false, false);
}

// ---------------------------------------------------------------------------
// feat f32 [256][4096] (NCHW) -> featT f16 [4096][256] (pixel-major) via LDS
// grid (64,4) x 256 threads; 64x64 tile, padded LDS to kill bank conflicts
// ---------------------------------------------------------------------------
__global__ void feat_to_f16(const float* __restrict__ feat,
                            _Float16* __restrict__ featT) {
  __shared__ float tile[64][65];
  const int p0 = blockIdx.x * 64, c0 = blockIdx.y * 64;
  const int tid = threadIdx.x;
#pragma unroll
  for (int i = 0; i < 16; ++i) {
    int row = i * 4 + (tid >> 6);   // channel offset
    int col = tid & 63;             // pixel offset (coalesced read)
    tile[row][col] = feat[(size_t)(c0 + row) * 4096 + p0 + col];
  }
  __syncthreads();
#pragma unroll
  for (int i = 0; i < 16; ++i) {
    int prow = i * 4 + (tid >> 6);
    int c = tid & 63;
    featT[(size_t)(p0 + prow) * 256 + c0 + c] = (_Float16)tile[c][prow];
  }
}

// ---------------------------------------------------------------------------
// Weight reorder: OIHW f32 -> [O][ky][kx][I] f16 (K-major for implicit GEMM)
// ---------------------------------------------------------------------------
__global__ void wprep(const float* __restrict__ src, _Float16* __restrict__ dst,
                      int O, int I, int KH, int KW) {
  int idx = blockIdx.x * 256 + threadIdx.x;
  int total = O * I * KH * KW;
  if (idx >= total) return;
  int ci = idx % I;
  int t = idx / I;
  int dx = t % KW; t /= KW;
  int dy = t % KH;
  int o  = t / KH;
  dst[idx] = (_Float16)src[((size_t)(o * I + ci) * KH + dy) * KW + dx];
}

// ---------------------------------------------------------------------------
// 1x1 projections: x1/x2 [p][c] f16 from featT f16, W f32 (K=256, tiny GEMM)
// one block per pixel; feat column staged in LDS; W rows stream from L2
// ---------------------------------------------------------------------------
__global__ void proj_kernel(const _Float16* __restrict__ featT,
                            const float* __restrict__ w1, const float* __restrict__ b1,
                            const float* __restrict__ w2, const float* __restrict__ b2,
                            _Float16* __restrict__ x1T, _Float16* __restrict__ x2T) {
  __shared__ float fp[256];
  const int p = blockIdx.x, c = threadIdx.x;
  fp[c] = (float)featT[(size_t)p * 256 + c];
  __syncthreads();
  const float* w = (c < 128) ? (w1 + (size_t)c * 256) : (w2 + (size_t)(c - 128) * 256);
  float acc = (c < 128) ? b1[c] : b2[c - 128];
#pragma unroll 8
  for (int k = 0; k < 256; ++k) acc += w[k] * fp[k];
  _Float16 h = (_Float16)acc;
  if (c < 128) x1T[(size_t)p * 128 + c] = h;
  else         x2T[(size_t)p * 128 + (c - 128)] = h;
}

// ---------------------------------------------------------------------------
// Score GEMM: S = sigmoid((x1T * x2T^T) / sqrt(128)), M=N=4096, K=128
// 1 wave = 32x32 tile (2x2 accumulators), 4 K-steps, 16 WMMAs per wave.
// A/B stream from L2 (2 MB total); output 64 MB -> HBM-store bound (~3us).
// ---------------------------------------------------------------------------
__global__ void score_gemm(const _Float16* __restrict__ x1T,
                           const _Float16* __restrict__ x2T,
                           float* __restrict__ out) {
  const int lane = threadIdx.x & 31;
  const int wid  = threadIdx.x >> 5;
  const int tile = blockIdx.x * 8 + wid;     // 16384 tiles exactly
  const int m0 = (tile >> 7) * 32;
  const int n0 = (tile & 127) * 32;
  const int half = lane & 15, hi = lane >> 4;

  v8f c00 = {}, c01 = {}, c10 = {}, c11 = {};
#pragma unroll
  for (int k0 = 0; k0 < 128; k0 += 32) {
    v16h a0 = ldA(x1T + (size_t)(m0 + half) * 128 + k0 + hi * 8);
    v16h a1 = ldA(x1T + (size_t)(m0 + 16 + half) * 128 + k0 + hi * 8);
    v16h b0 = ldB(x2T + (size_t)(n0 + half) * 128 + k0 + hi * 8);
    v16h b1 = ldB(x2T + (size_t)(n0 + 16 + half) * 128 + k0 + hi * 8);
    c00 = wmma_f16(a0, b0, c00);
    c01 = wmma_f16(a0, b1, c01);
    c10 = wmma_f16(a1, b0, c10);
    c11 = wmma_f16(a1, b1, c11);
  }
  const float sc = 0.08838834764831845f;  // 1/sqrt(128)
  const int nA = n0 + half, nB = n0 + 16 + half;
#pragma unroll
  for (int r = 0; r < 8; ++r) {
    int mA = m0 + hi * 8 + r;
    int mB = m0 + 16 + hi * 8 + r;
    out[(size_t)mA * 4096 + nA] = 1.f / (1.f + expf(-c00[r] * sc));
    out[(size_t)mA * 4096 + nB] = 1.f / (1.f + expf(-c01[r] * sc));
    out[(size_t)mB * 4096 + nA] = 1.f / (1.f + expf(-c10[r] * sc));
    out[(size_t)mB * 4096 + nB] = 1.f / (1.f + expf(-c11[r] * sc));
  }
}

// ---------------------------------------------------------------------------
// Implicit-GEMM conv (KHxKW, pad, stride) + fused BN (+residual) (+ReLU).
// inT:  [IH*IW][Cin] f16    wr: [Cout][KH*KW*Cin] f16    outT: [OH*OW][Cout]
// 1 wave = 32(M=Cout) x 32(N=pixels) tile; B fragment = one v16h load per
// (tap, 32-chan) step with zero-fill at padded borders.
// ---------------------------------------------------------------------------
__global__ void conv_wmma(const _Float16* __restrict__ inT, int IH, int IW, int Cin,
                          const _Float16* __restrict__ wr, int Cout,
                          const float* __restrict__ bns, const float* __restrict__ bnb,
                          const _Float16* __restrict__ resid, int do_relu,
                          _Float16* __restrict__ outT, int OH, int OW,
                          int stride, int pad, int KH, int KW) {
  const int lane = threadIdx.x & 31;
  const int wid  = threadIdx.x >> 5;
  const int N  = OH * OW;
  const int nt = N >> 5;
  const int tiles = (Cout >> 5) * nt;
  const int tile = blockIdx.x * 8 + wid;
  if (tile >= tiles) return;                 // wave-uniform: EXEC stays all-1s
  const int m0 = (tile / nt) * 32;
  const int n0 = (tile % nt) * 32;
  const int half = lane & 15, hi = lane >> 4;
  const int K = KH * KW * Cin;

  const int nA = n0 + half, nB = n0 + 16 + half;  // per-lane output pixels
  const int oyA = nA / OW, oxA = nA % OW;
  const int oyB = nB / OW, oxB = nB % OW;

  const _Float16* arow0 = wr + (size_t)(m0 + half) * K + hi * 8;
  const _Float16* arow1 = wr + (size_t)(m0 + 16 + half) * K + hi * 8;

  v8f c00 = {}, c01 = {}, c10 = {}, c11 = {};
  const v16h vz = {};

  for (int t = 0; t < KH * KW; ++t) {
    const int dy = t / KW, dx = t % KW;
    const int iyA = oyA * stride + dy - pad, ixA = oxA * stride + dx - pad;
    const int iyB = oyB * stride + dy - pad, ixB = oxB * stride + dx - pad;
    const bool inA = (iyA >= 0) & (iyA < IH) & (ixA >= 0) & (ixA < IW);
    const bool inB = (iyB >= 0) & (iyB < IH) & (ixB >= 0) & (ixB < IW);
    const _Float16* bA = inT + (size_t)(iyA * IW + ixA) * Cin + hi * 8;
    const _Float16* bB = inT + (size_t)(iyB * IW + ixB) * Cin + hi * 8;
    const int kbase = t * Cin;
    for (int ci = 0; ci < Cin; ci += 32) {
      v16h a0 = ldA(arow0 + kbase + ci);
      v16h a1 = ldA(arow1 + kbase + ci);
      v16h b0 = inA ? ldB(bA + ci) : vz;     // predicate only gates the loads;
      v16h b1 = inB ? ldB(bB + ci) : vz;     // WMMA runs with full EXEC
      c00 = wmma_f16(a0, b0, c00);
      c01 = wmma_f16(a0, b1, c01);
      c10 = wmma_f16(a1, b0, c10);
      c11 = wmma_f16(a1, b1, c11);
    }
  }

  // epilogue: BN (+residual) (+ReLU), 16B v8h store per fragment
  auto store8 = [&](int n, int mbase, v8f d) {
    v8h o;
#pragma unroll
    for (int r = 0; r < 8; ++r) {
      int m = mbase + r;
      float y = d[r] * bns[m] + bnb[m];
      if (resid) y += (float)resid[(size_t)n * Cout + m];
      if (do_relu) y = y > 0.f ? y : 0.f;
      o[r] = (_Float16)y;
    }
    *(v8h*)(outT + (size_t)n * Cout + mbase) = o;
  };
  store8(nA, m0 + hi * 8,        c00);
  store8(nB, m0 + hi * 8,        c01);
  store8(nA, m0 + 16 + hi * 8,   c10);
  store8(nB, m0 + 16 + hi * 8,   c11);
}

// ---------------------------------------------------------------------------
// Global avg-pool + FC(512->2) + softmax
// ---------------------------------------------------------------------------
__global__ void head_kernel(const _Float16* __restrict__ a3,
                            const float* __restrict__ fcw,
                            const float* __restrict__ fcb,
                            float* __restrict__ outLabel) {
  __shared__ float pooled[512];
  __shared__ float logit[2];
  const int c = threadIdx.x;  // 512 threads
  float s = 0.f;
  for (int p = 0; p < 1024; ++p) s += (float)a3[(size_t)p * 512 + c];
  pooled[c] = s * (1.f / 1024.f);
  __syncthreads();
  if (c < 2) {
    float acc = fcb[c];
    for (int k = 0; k < 512; ++k) acc += fcw[c * 512 + k] * pooled[k];
    logit[c] = acc;
  }
  __syncthreads();
  if (c == 0) {
    float m = fmaxf(logit[0], logit[1]);
    float e0 = expf(logit[0] - m), e1 = expf(logit[1] - m);
    float inv = 1.f / (e0 + e1);
    outLabel[0] = e0 * inv;
    outLabel[1] = e1 * inv;
  }
}

// ---------------------------------------------------------------------------
extern "C" void kernel_launch(void* const* d_in, const int* in_sizes, int n_in,
                              void* d_out, int out_size, void* d_ws, size_t ws_size,
                              hipStream_t stream) {
  const float* feat  = (const float*)d_in[0];
  const float* w1    = (const float*)d_in[1];
  const float* b1    = (const float*)d_in[2];
  const float* w2    = (const float*)d_in[3];
  const float* b2    = (const float*)d_in[4];
  const float* c1w[6] = {(const float*)d_in[5],  (const float*)d_in[8],
                         (const float*)d_in[14], (const float*)d_in[17],
                         (const float*)d_in[20], (const float*)d_in[23]};
  const float* bnS[6] = {(const float*)d_in[6],  (const float*)d_in[9],
                         (const float*)d_in[15], (const float*)d_in[18],
                         (const float*)d_in[21], (const float*)d_in[24]};
  const float* bnB[6] = {(const float*)d_in[7],  (const float*)d_in[10],
                         (const float*)d_in[16], (const float*)d_in[19],
                         (const float*)d_in[22], (const float*)d_in[25]};
  const float* dw  = (const float*)d_in[11];
  const float* ds  = (const float*)d_in[12];
  const float* db  = (const float*)d_in[13];
  const float* fcw = (const float*)d_in[26];
  const float* fcb = (const float*)d_in[27];

  float* out = (float*)d_out;                 // [0,16777216) score, then label[2]
  char* ws = (char*)d_ws;                      // needs ~36 MB

  size_t off = 0;
  auto alloc = [&](size_t bytes) { char* p = ws + off; off = (off + bytes + 255) & ~size_t(255); return p; };
  _Float16* featT = (_Float16*)alloc((size_t)4096 * 256 * 2);
  _Float16* x1T   = (_Float16*)alloc((size_t)4096 * 128 * 2);
  _Float16* x2T   = (_Float16*)alloc((size_t)4096 * 128 * 2);
  _Float16* wrc[6];
  const int cinL[6] = {256, 512, 512, 512, 512, 512};
  for (int i = 0; i < 6; ++i) wrc[i] = (_Float16*)alloc((size_t)512 * 9 * cinL[i] * 2);
  _Float16* wrds = (_Float16*)alloc((size_t)512 * 256 * 2);
  _Float16* hbuf = (_Float16*)alloc((size_t)1024 * 512 * 2);
  _Float16* scb  = (_Float16*)alloc((size_t)1024 * 512 * 2);
  _Float16* a1   = (_Float16*)alloc((size_t)1024 * 512 * 2);
  _Float16* a2   = (_Float16*)alloc((size_t)1024 * 512 * 2);
  _Float16* a3   = (_Float16*)alloc((size_t)1024 * 512 * 2);
  (void)ws_size; (void)in_sizes; (void)n_in; (void)out_size;

  // --- data prep -----------------------------------------------------------
  feat_to_f16<<<dim3(64, 4), 256, 0, stream>>>(feat, featT);
  for (int i = 0; i < 6; ++i) {
    int total = 512 * cinL[i] * 9;
    wprep<<<(total + 255) / 256, 256, 0, stream>>>(c1w[i], wrc[i], 512, cinL[i], 3, 3);
  }
  wprep<<<(512 * 256 + 255) / 256, 256, 0, stream>>>(dw, wrds, 512, 256, 1, 1);

  // --- branch 1: projections + score volume (WMMA GEMM, 64 MB store) -------
  proj_kernel<<<4096, 256, 0, stream>>>(featT, w1, b1, w2, b2, x1T, x2T);
  score_gemm<<<2048, 256, 0, stream>>>(x1T, x2T, out);

  // --- branch 2: ResNet layer4 via WMMA implicit-GEMM convs ----------------
  const int convBlocks = ((512 / 32) * (1024 / 32) + 7) / 8;  // 64
  // block0: conv1 (stride 2) -> h ; downsample (1x1 s2) -> sc ; conv2 + sc -> a1
  conv_wmma<<<convBlocks, 256, 0, stream>>>(featT, 64, 64, 256, wrc[0], 512,
      bnS[0], bnB[0], nullptr, 1, hbuf, 32, 32, 2, 1, 3, 3);
  conv_wmma<<<convBlocks, 256, 0, stream>>>(featT, 64, 64, 256, wrds, 512,
      ds, db, nullptr, 0, scb, 32, 32, 2, 0, 1, 1);
  conv_wmma<<<convBlocks, 256, 0, stream>>>(hbuf, 32, 32, 512, wrc[1], 512,
      bnS[1], bnB[1], scb, 1, a1, 32, 32, 1, 1, 3, 3);
  // block1
  conv_wmma<<<convBlocks, 256, 0, stream>>>(a1, 32, 32, 512, wrc[2], 512,
      bnS[2], bnB[2], nullptr, 1, hbuf, 32, 32, 1, 1, 3, 3);
  conv_wmma<<<convBlocks, 256, 0, stream>>>(hbuf, 32, 32, 512, wrc[3], 512,
      bnS[3], bnB[3], a1, 1, a2, 32, 32, 1, 1, 3, 3);
  // block2
  conv_wmma<<<convBlocks, 256, 0, stream>>>(a2, 32, 32, 512, wrc[4], 512,
      bnS[4], bnB[4], nullptr, 1, hbuf, 32, 32, 1, 1, 3, 3);
  conv_wmma<<<convBlocks, 256, 0, stream>>>(hbuf, 32, 32, 512, wrc[5], 512,
      bnS[5], bnB[5], a2, 1, a3, 32, 32, 1, 1, 3, 3);

  // --- head ----------------------------------------------------------------
  head_kernel<<<1, 512, 0, stream>>>(a3, fcw, fcb, out + (size_t)16777216);
}
